// MultiHeadAttention_85383949844845
// MI455X (gfx1250) — matrix operand order repaired and checked by reading, not compiled, not run on hardware
//
#include <hip/hip_runtime.h>

typedef _Float16 v16h __attribute__((ext_vector_type(16)));
typedef _Float16 v8h  __attribute__((ext_vector_type(8)));
typedef float    v8f  __attribute__((ext_vector_type(8)));

#define B_   8
#define LQ_  2048
#define LK_  2048
#define D_   128
#define H_   8
#define HD_  1024   // H*D

// ---- CDNA5 async global->LDS copy (ASYNCcnt-tracked), with safe fallback ---
#if defined(__has_builtin)
#if __has_builtin(__builtin_amdgcn_global_load_async_to_lds_b128) && \
    __has_builtin(__builtin_amdgcn_s_wait_asynccnt)
#define USE_ASYNC_LDS 1
#endif
#endif
#ifndef USE_ASYNC_LDS
#define USE_ASYNC_LDS 0
#endif

#if USE_ASYNC_LDS
typedef int v4i_vs __attribute__((vector_size(4 * sizeof(int))));
typedef __attribute__((address_space(1))) v4i_vs* as1_v4i;
typedef __attribute__((address_space(3))) v4i_vs* as3_v4i;
#endif

__device__ __forceinline__ void cp_async16(const _Float16* g, _Float16* l) {
#if USE_ASYNC_LDS
  __builtin_amdgcn_global_load_async_to_lds_b128(
      (as1_v4i)(const_cast<_Float16*>(g)), (as3_v4i)l, 0, 0);
#else
  *(uint4*)l = *(const uint4*)g;
#endif
}

__device__ __forceinline__ void wait_async_done() {
#if USE_ASYNC_LDS
  __builtin_amdgcn_s_wait_asynccnt(0);
#endif
}

__device__ __forceinline__ v8f wmma_f16(v16h a, v16h b, v8f c) {
  // D = A(16x32 f16) * B(32x16 f16) + C(16x16 f32)
  return __builtin_amdgcn_wmma_f32_16x16x32_f16(false, a, false, b, (short)0, c,
                                                false, false);
}

// ---- WMMA fragment loaders (wave32; contiguous k = 16*hi + s in both A and B) ----------

// A-fragment (16x32, f16) from row-major f16 tile, ld in halfs.
__device__ __forceinline__ v16h load_afrag(const _Float16* base, int ld) {
  int lane = threadIdx.x & 31;
  int r = lane & 15, hi = (lane >> 4) & 1;
  const _Float16* p = base + (size_t)r * ld + hi * 16;
  v8h c0 = *(const v8h*)(p);
  v8h c1 = *(const v8h*)(p + 8);
  v16h a;
#pragma unroll
  for (int i = 0; i < 8; ++i) { a[i] = c0[i]; a[i + 8] = c1[i]; }
  return a;
}

// B-fragment (32x16, f16) from a source laid out [n][k].
__device__ __forceinline__ v16h load_bfrag(const _Float16* base, int ld) {
  int lane = threadIdx.x & 31;
  int r = lane & 15, hi = (lane >> 4) & 1;
  const _Float16* p = base + (size_t)r * ld + hi * 16;
  v8h c0 = *(const v8h*)(p);
  v8h c1 = *(const v8h*)(p + 8);
  v16h b;
#pragma unroll
  for (int i = 0; i < 8; ++i) { b[i] = c0[i]; b[i + 8] = c1[i]; }
  return b;
}

// A-fragment from row-major f32 source, converted to f16.
__device__ __forceinline__ v16h load_afrag_f32(const float* base, int ld) {
  int lane = threadIdx.x & 31;
  int r = lane & 15, hi = (lane >> 4) & 1;
  const float* p = base + (size_t)r * ld + hi * 16;
  v16h a;
#pragma unroll
  for (int i = 0; i < 8; ++i) {
    a[i]     = (_Float16)p[i];
    a[i + 8] = (_Float16)p[i + 8];
  }
  return a;
}

// B-fragment from f32 weight matrix W[K][N] (row-major, stride N).
__device__ __forceinline__ v16h load_bfrag_w(const float* W, int N, int k0, int n0) {
  int lane = threadIdx.x & 31;
  int n = n0 + (lane & 15);
  int kb = k0 + ((lane >> 4) & 1) * 16;
  v16h b;
#pragma unroll
  for (int e = 0; e < 16; ++e) b[e] = (_Float16)W[(size_t)(kb + e) * N + n];
  return b;
}

// ---- kernel 1: input projections --------------------------------------------
// X[B*L,128]f32 @ W[128,1024]f32 + bias -> f16
// mode 0: out[b][h][l][d]   (Q/K layout; Q pre-scaled by 1/sqrt(d))
// mode 1: out[b][h][d][l]   (V transposed layout)
__global__ __launch_bounds__(128) void proj_kernel(const float* __restrict__ X,
                                                   const float* __restrict__ W,
                                                   const float* __restrict__ bias,
                                                   _Float16* __restrict__ out,
                                                   int mode, float scale) {
  int wave = threadIdx.x >> 5;
  int lane = threadIdx.x & 31;
  int r16 = lane & 15, hi = (lane >> 4) & 1;
  int tile = blockIdx.x * 4 + wave;   // 65536 tiles
  int tm = tile >> 6;                 // 1024 row tiles
  int tn = tile & 63;                 // 64 col tiles

  v8f acc;
#pragma unroll
  for (int i = 0; i < 8; ++i) acc[i] = 0.f;

#pragma unroll
  for (int s = 0; s < 4; ++s) {
    v16h a = load_afrag_f32(X + (size_t)tm * 16 * D_ + s * 32, D_);
    v16h b = load_bfrag_w(W, HD_, s * 32, tn * 16);
    acc = wmma_f16(a, b, acc);
  }

  int n = tn * 16 + r16;
  float bn = bias[n];
  int h = n >> 7, dn = n & 127;

  if (mode == 0) {
#pragma unroll
    for (int r = 0; r < 8; ++r) {
      int M = tm * 16 + r + 8 * hi;
      int bidx = M >> 11;
      int l = M & 2047;
      out[(((size_t)bidx * H_ + h) * LQ_ + l) * D_ + dn] =
          (_Float16)((acc[r] + bn) * scale);
    }
  } else {
    int M0 = tm * 16 + 8 * hi;
    int bidx = M0 >> 11;
    int l0 = M0 & 2047;
    v8h pk;
#pragma unroll
    for (int r = 0; r < 8; ++r) pk[r] = (_Float16)((acc[r] + bn) * scale);
    *(v8h*)(out + (((size_t)bidx * H_ + h) * D_ + dn) * (size_t)LK_ + l0) = pk;
  }
}

// ---- kernel 2: flash attention (8 waves, 128 q-rows/block, double-buffered) -
__global__ __launch_bounds__(256) void attn_kernel(const _Float16* __restrict__ Qh,
                                                   const _Float16* __restrict__ Kh,
                                                   const _Float16* __restrict__ VhT,
                                                   _Float16* __restrict__ Oh) {
  __shared__ __align__(16) _Float16 Kt[2][64][136];   // 64 keys x 128d (+pad)
  __shared__ __align__(16) _Float16 Vt[2][128][72];   // 128 dv x 64 keys (+pad)
  __shared__ __align__(16) _Float16 Pt[8][16][72];    // per-wave P staging

  int tid = threadIdx.x;
  int wave = tid >> 5, lane = tid & 31;
  int r16 = lane & 15, hi = (lane >> 4) & 1;

  int blk = blockIdx.x;
  int qb = blk & 15;    // q-block of 128 rows
  int bh = blk >> 4;    // b*H + h

  const _Float16* Qbase = Qh + ((size_t)bh * LQ_ + qb * 128 + wave * 16) * D_;
  const _Float16* Kbase = Kh + (size_t)bh * LK_ * D_;
  const _Float16* Vbase = VhT + (size_t)bh * D_ * LK_;

  // issue async copies of k-block `kb` into buffer `buf`
  auto issue_tile = [&](int buf, int kb) {
    {  // K tile: 64 rows x 128 halfs; 256 threads -> 4x16B each
      int row = tid >> 2, seg = tid & 3;
      const _Float16* g = Kbase + ((size_t)(kb * 64 + row)) * D_ + seg * 32;
      _Float16* l = &Kt[buf][row][seg * 32];
#pragma unroll
      for (int i = 0; i < 4; ++i) cp_async16(g + i * 8, l + i * 8);
    }
    {  // V^T tile: 128 rows x 64 halfs; 256 threads -> 4x16B each
      int row = tid >> 1, seg = tid & 1;
      const _Float16* g = Vbase + (size_t)row * LK_ + kb * 64 + seg * 32;
      _Float16* l = &Vt[buf][row][seg * 32];
#pragma unroll
      for (int i = 0; i < 4; ++i) cp_async16(g + i * 8, l + i * 8);
    }
  };

  // Q fragments for this wave's 16 rows (d=128 in 4 K-slices of 32)
  v16h aq[4];
#pragma unroll
  for (int s = 0; s < 4; ++s) aq[s] = load_afrag(Qbase + s * 32, D_);

  v8f o[8];
  float m_i[8], l_i[8];
#pragma unroll
  for (int t = 0; t < 8; ++t)
#pragma unroll
    for (int r = 0; r < 8; ++r) o[t][r] = 0.f;
#pragma unroll
  for (int r = 0; r < 8; ++r) { m_i[r] = -1e30f; l_i[r] = 0.f; }

  issue_tile(0, 0);

  for (int kb = 0; kb < LK_ / 64; ++kb) {
    int cur = kb & 1;
    wait_async_done();     // this wave's async copies landed in LDS
    __syncthreads();       // ... and everyone else's too
    if (kb + 1 < LK_ / 64) issue_tile(cur ^ 1, kb + 1);

    // S = Q K^T : 4 key-tiles of 16, reduce over d in 4 slices of 32
    v8f sc[4];
#pragma unroll
    for (int nt = 0; nt < 4; ++nt) {
#pragma unroll
      for (int r = 0; r < 8; ++r) sc[nt][r] = 0.f;
#pragma unroll
      for (int ks = 0; ks < 4; ++ks) {
        v16h bk = load_bfrag(&Kt[cur][nt * 16][ks * 32], 136);
        sc[nt] = wmma_f16(aq[ks], bk, sc[nt]);
      }
    }

    // online softmax (row stats via 16-lane xor reductions)
    float corr[8];
#pragma unroll
    for (int r = 0; r < 8; ++r) {
      float v = fmaxf(fmaxf(sc[0][r], sc[1][r]), fmaxf(sc[2][r], sc[3][r]));
#pragma unroll
      for (int m = 8; m >= 1; m >>= 1) v = fmaxf(v, __shfl_xor(v, m, 32));
      float mnew = fmaxf(m_i[r], v);
      corr[r] = __expf(m_i[r] - mnew);
      float rs = 0.f;
#pragma unroll
      for (int nt = 0; nt < 4; ++nt) {
        float p = __expf(sc[nt][r] - mnew);
        sc[nt][r] = p;
        rs += p;
      }
#pragma unroll
      for (int m = 8; m >= 1; m >>= 1) rs += __shfl_xor(rs, m, 32);
      l_i[r] = l_i[r] * corr[r] + rs;
      m_i[r] = mnew;
    }
#pragma unroll
    for (int t = 0; t < 8; ++t)
#pragma unroll
      for (int r = 0; r < 8; ++r) o[t][r] *= corr[r];

    // stage P (C-layout -> LDS) so it can be re-read in A-fragment layout
#pragma unroll
    for (int nt = 0; nt < 4; ++nt)
#pragma unroll
      for (int r = 0; r < 8; ++r)
        Pt[wave][r + 8 * hi][nt * 16 + r16] = (_Float16)sc[nt][r];

    // O += P V : reduce over 64 keys in 2 slices of 32
#pragma unroll
    for (int ks = 0; ks < 2; ++ks) {
      v16h ap = load_afrag(&Pt[wave][0][ks * 32], 72);
#pragma unroll
      for (int dt = 0; dt < 8; ++dt) {
        v16h bv = load_bfrag(&Vt[cur][dt * 16][ks * 32], 72);
        o[dt] = wmma_f16(ap, bv, o[dt]);
      }
    }
  }

  // epilogue: normalize and store Oh[b][lq][h][dv] as f16
  int b = bh >> 3, h = bh & 7;
#pragma unroll
  for (int r = 0; r < 8; ++r) {
    float inv = 1.0f / l_i[r];
    int lq = qb * 128 + wave * 16 + r + 8 * hi;
    size_t rowoff = (((size_t)b * LQ_ + lq) * H_ + h) * D_;
#pragma unroll
    for (int dt = 0; dt < 8; ++dt)
      Oh[rowoff + dt * 16 + r16] = (_Float16)(o[dt][r] * inv);
  }
}

// ---- kernel 3: output projection -------------------------------------------
// Oh[16384,1024]f16 @ Wo[1024,128]f32 + bo -> out f32
__global__ __launch_bounds__(128) void outproj_kernel(const _Float16* __restrict__ Oh,
                                                      const float* __restrict__ Wo,
                                                      const float* __restrict__ bo,
                                                      float* __restrict__ out) {
  int wave = threadIdx.x >> 5, lane = threadIdx.x & 31;
  int r16 = lane & 15, hi = (lane >> 4) & 1;
  int tile = blockIdx.x * 4 + wave;   // 8192 tiles
  int tm = tile >> 3;                 // 1024 row tiles
  int tn = tile & 7;                  // 8 col tiles

  v8f acc;
#pragma unroll
  for (int i = 0; i < 8; ++i) acc[i] = 0.f;

  const _Float16* Abase = Oh + (size_t)tm * 16 * HD_;
#pragma unroll 4
  for (int s = 0; s < 32; ++s) {
    v16h a = load_afrag(Abase + s * 32, HD_);
    v16h b = load_bfrag_w(Wo, D_, s * 32, tn * 16);
    acc = wmma_f16(a, b, acc);
  }

  int n = tn * 16 + r16;
  float bn = bo[n];
#pragma unroll
  for (int r = 0; r < 8; ++r) {
    int M = tm * 16 + r + 8 * hi;
    out[(size_t)M * D_ + n] = acc[r] + bn;
  }
}

// ---- host launch ------------------------------------------------------------
extern "C" void kernel_launch(void* const* d_in, const int* in_sizes, int n_in,
                              void* d_out, int out_size, void* d_ws, size_t ws_size,
                              hipStream_t stream) {
  (void)in_sizes; (void)n_in; (void)out_size; (void)ws_size;
  const float* q  = (const float*)d_in[0];
  const float* k  = (const float*)d_in[1];
  const float* v  = (const float*)d_in[2];
  const float* Wq = (const float*)d_in[3];
  const float* bq = (const float*)d_in[4];
  const float* Wk = (const float*)d_in[5];
  const float* bk = (const float*)d_in[6];
  const float* Wv = (const float*)d_in[7];
  const float* bv = (const float*)d_in[8];
  const float* Wo = (const float*)d_in[9];
  const float* bo = (const float*)d_in[10];

  const size_t NH = (size_t)B_ * H_ * LQ_ * D_;  // 16M halfs = 32 MB each
  _Float16* Qh  = (_Float16*)d_ws;
  _Float16* Kh  = Qh + NH;
  _Float16* VhT = Kh + NH;
  _Float16* Oh  = VhT + NH;

  const float scale = 0.08838834764831845f;  // 1/sqrt(128)
  proj_kernel<<<16384, 128, 0, stream>>>(q, Wq, bq, Qh, 0, scale);
  proj_kernel<<<16384, 128, 0, stream>>>(k, Wk, bk, Kh, 0, 1.0f);
  proj_kernel<<<16384, 128, 0, stream>>>(v, Wv, bv, VhT, 1, 1.0f);
  attn_kernel<<<B_ * H_ * (LQ_ / 128), 256, 0, stream>>>(Qh, Kh, VhT, Oh);
  outproj_kernel<<<2048, 128, 0, stream>>>(Oh, Wo, bo, (float*)d_out);
}
